// LSTMModel_64604898066544
// MI455X (gfx1250) — matrix operand order repaired and checked by reading, not compiled, not run on hardware
//
#include <hip/hip_runtime.h>
#include <hip/hip_bf16.h>

// Problem dims (fixed by reference)
#define V_ 32000
#define H_ 1024
#define L_ 4
#define S_ 256
#define B_ 16

// LDS row layout for staged 16x1024 bf16 tiles:
// 2048B of data per row + 16B pad after every 1024B (TDM pad_interval=256dw, pad_amount=4dw)
// -> row stride 2080B, element k at k*2 + (k>=512 ? 16 : 0). 16B alignment preserved.
#define SROW 2080

typedef __bf16 bf16_t;
typedef __attribute__((ext_vector_type(16))) __bf16 v16bf;
typedef __attribute__((ext_vector_type(8)))  __bf16 v8bf;
typedef __attribute__((ext_vector_type(8)))  float  v8f;

typedef unsigned int u32x4 __attribute__((ext_vector_type(4)));
typedef int          i32x4 __attribute__((ext_vector_type(4)));
typedef int          i32x8 __attribute__((ext_vector_type(8)));

// ---------- scalar helpers ----------

__device__ __forceinline__ bf16_t f2bf(float f) {
  union { float f; unsigned u; } in; in.f = f;
  unsigned u = in.u;
  unsigned r = u + 0x7fffu + ((u >> 16) & 1u);   // round-to-nearest-even
  union { unsigned short s; bf16_t b; } out;
  out.s = (unsigned short)(r >> 16);
  return out.b;
}

__device__ __forceinline__ float sigmoidf_(float x) {
  return 1.0f / (1.0f + __expf(-x));
}

// ---------- tile loaders ----------

// A-matrix 16x32 bf16 tile from padded LDS buffer (contiguous k order, same as load_b32x16):
//   lane&15 -> M row; lane<16: K 0..15; lane>=16: K 16..31 (contiguous)
__device__ __forceinline__ v16bf lds_a16x32(const char* sbase, int k0) {
  const int lane = threadIdx.x & 31;
  const int m    = lane & 15;
  const int koff = (lane & 16) ? 16 : 0;
  const int k1 = k0 + koff;
  const int k2 = k1 + 8;
  const char* row = sbase + m * SROW;
  v8bf lo = *(const v8bf*)(row + k1 * 2 + ((k1 >= 512) ? 16 : 0));
  v8bf hi = *(const v8bf*)(row + k2 * 2 + ((k2 >= 512) ? 16 : 0));
  v16bf a;
#pragma unroll
  for (int i = 0; i < 8; ++i) { a[i] = lo[i]; a[8 + i] = hi[i]; }
  return a;
}

// B-matrix 32x16 bf16 tile from row-major W[N,K] (B[k][n] = W[n0+n][k0+k]):
//   lane&15 -> N column; lane<16 holds K=0..15, lane>=16 holds K=16..31 (contig 32B)
__device__ __forceinline__ v16bf load_b32x16(const bf16_t* __restrict__ W,
                                             int n0, int ldk, int k0) {
  const int lane = threadIdx.x & 31;
  const int n    = lane & 15;
  const int kh   = (lane & 16) ? 16 : 0;
  return *(const v16bf*)(W + (size_t)(n0 + n) * ldk + k0 + kh);
}

__device__ __forceinline__ v8f wmma_bf16(v16bf a, v16bf b, v8f c) {
  return __builtin_amdgcn_wmma_f32_16x16x32_bf16(
      false, a, false, b, (short)0, c, false, false);
}

// ---------- global -> LDS staging of one 16x1024 bf16 tile ----------

#if __has_builtin(__builtin_amdgcn_tensor_load_to_lds)
// Tensor Data Mover: one descriptor moves the whole 32KB tile, HW inserts the
// 16B LDS pad every 1024B so ds_load A-tile reads are bank-spread.
// This toolchain exposes the 6-arg builtin: (g0, g1, g2, g3, g4, cpol).
__device__ __forceinline__ void tdm_load_16x1024(void* s, const void* g) {
  unsigned long long ga = (unsigned long long)g;
  // gfx1250: flat pointer to LDS keeps the byte offset in the low 32 bits
  unsigned lds_off = (unsigned)(unsigned long long)s;
  u32x4 g0;
  g0[0] = 1u;                                            // count=1 (valid)
  g0[1] = lds_off;                                       // lds_addr
  g0[2] = (unsigned)(ga & 0xFFFFFFFFu);                  // global_addr[31:0]
  g0[3] = (unsigned)((ga >> 32) & 0x01FFFFFFu) | (2u << 30);  // addr[56:32], type=2
  i32x8 g1;
  g1[0] = (1 << 16)        // data_size = 2 bytes
        | (1 << 20)        // pad_enable
        | (7 << 22)        // pad_interval = 256 dwords (1024B)
        | (3 << 25);       // pad_amount   = 4 dwords (16B)
  g1[1] = (int)((1024u & 0xFFFFu) << 16);                // tensor_dim0 lo16
  g1[2] = (int)(((1024u >> 16) & 0xFFFFu) | (16u << 16));// tensor_dim0 hi / dim1 lo
  g1[3] = (int)(((16u >> 16) & 0xFFFFu) | (1024u << 16));// dim1 hi / tile_dim0
  g1[4] = (int)16;                                       // tile_dim1, tile_dim2=0
  g1[5] = (int)1024;                                     // tensor_dim0_stride lo32
  g1[6] = 0;
  g1[7] = 0;
  i32x4 z4 = {0, 0, 0, 0};
  i32x8 z8 = {0, 0, 0, 0, 0, 0, 0, 0};
  __builtin_amdgcn_tensor_load_to_lds(g0, g1, z4, z4, z8, 0);
}
#else
// Fallback: cooperative vector copy into the same padded layout.
__device__ __forceinline__ void coop_stage_16x1024(char* s, const bf16_t* g) {
  for (int idx = threadIdx.x; idx < 16 * 128; idx += blockDim.x) {
    int m = idx >> 7;
    int c = (idx & 127) << 3;                            // chunk start k
    v8bf v = *(const v8bf*)(g + (size_t)m * 1024 + c);
    *(v8bf*)(s + m * SROW + c * 2 + ((c >= 512) ? 16 : 0)) = v;
  }
}
#endif

// ---------- LSTM cell: one step of one layer ----------
// grid = 8 blocks x 256 threads = 64 waves; wave w owns hidden units [16w,16w+16)
__global__ void __launch_bounds__(256)
lstm_cell_kernel(const bf16_t* __restrict__ xin,    // [B,H] bf16 layer input at t
                 const bf16_t* __restrict__ hprev,  // [B,H] bf16 h_{t-1}
                 const bf16_t* __restrict__ Wih,    // [4H,H] bf16
                 const bf16_t* __restrict__ Whh,    // [4H,H] bf16
                 const float*  __restrict__ bias,   // [4H] = b_ih + b_hh
                 float*        __restrict__ c_st,   // [B,H] f32 (in/out)
                 float*        __restrict__ h_st,   // [B,H] f32 (out)
                 bf16_t*       __restrict__ hout)   // [B,H] bf16 (out)
{
  __shared__ __align__(128) char sX[16 * SROW];
  __shared__ __align__(128) char sH[16 * SROW];

#if __has_builtin(__builtin_amdgcn_tensor_load_to_lds)
  if ((threadIdx.x >> 5) == 0) {     // wave 0 issues both DMAs
    tdm_load_16x1024(sX, xin);
    tdm_load_16x1024(sH, hprev);
    __builtin_amdgcn_s_wait_tensorcnt(0);
  }
#else
  coop_stage_16x1024(sX, xin);
  coop_stage_16x1024(sH, hprev);
#endif
  __syncthreads();

  const int wave = blockIdx.x * (blockDim.x >> 5) + (threadIdx.x >> 5);
  const int u0   = wave << 4;        // hidden-unit tile base
  const int lane = threadIdx.x & 31;

  v8f acc[4] = {};                   // gates i,f,g,o

  // Software-pipelined: B tiles for stage s+1 are in flight while stage s does WMMA.
  v16bf bi = load_b32x16(Wih, u0, H_, 0);
  v16bf bh = load_b32x16(Whh, u0, H_, 0);
  for (int k0 = 0; k0 < H_; k0 += 32) {
    v16bf ax = lds_a16x32(sX, k0);
    v16bf ah = lds_a16x32(sH, k0);
#pragma unroll
    for (int g = 0; g < 4; ++g) {
      v16bf bi2, bh2;
      if (g < 3) {
        bi2 = load_b32x16(Wih, (g + 1) * H_ + u0, H_, k0);
        bh2 = load_b32x16(Whh, (g + 1) * H_ + u0, H_, k0);
      } else {
        const int kn = (k0 + 32) & (H_ - 1);   // wrap on last iter (harmless)
        bi2 = load_b32x16(Wih, u0, H_, kn);
        bh2 = load_b32x16(Whh, u0, H_, kn);
      }
      acc[g] = wmma_bf16(ax, bi, acc[g]);
      acc[g] = wmma_bf16(ah, bh, acc[g]);
      bi = bi2;
      bh = bh2;
    }
  }

  // Accumulator layout: VGPR r, lane -> (M = r + 8*(lane>=16), N = lane&15)
  const int   n   = lane & 15;
  const int   u   = u0 + n;
  const float bi_ = bias[u];
  const float bf_ = bias[H_ + u];
  const float bg_ = bias[2 * H_ + u];
  const float bo_ = bias[3 * H_ + u];
  const int   mb  = (lane & 16) ? 8 : 0;

#pragma unroll
  for (int r = 0; r < 8; ++r) {
    const int m = mb + r;
    float zi = acc[0][r] + bi_;
    float zf = acc[1][r] + bf_;
    float zg = acc[2][r] + bg_;
    float zo = acc[3][r] + bo_;
    size_t idx = (size_t)m * H_ + u;
    float cp = c_st[idx];
    float cn = sigmoidf_(zf) * cp + sigmoidf_(zi) * tanhf(zg);
    float hn = sigmoidf_(zo) * tanhf(cn);
    c_st[idx] = cn;
    h_st[idx] = hn;
    hout[idx] = f2bf(hn);
  }
}

// ---------- FC: logits[m, v] = y[m,:] . fcW[v,:] + fc_b[v] ----------
// grid = (M/16, 125) x 256 threads; block shares one A tile (LDS);
// each wave computes 2 adjacent N tiles so row stores form one 128B line.
__global__ void __launch_bounds__(256)
fc_gemm_kernel(const bf16_t* __restrict__ A,    // [4096, 1024] bf16
               const bf16_t* __restrict__ Wf,   // [32000, 1024] bf16
               const float*  __restrict__ fb,   // [32000]
               float*        __restrict__ out)  // [4096, 32000]
{
  __shared__ __align__(128) char sA[16 * SROW];

  const int m0 = blockIdx.x << 4;

#if __has_builtin(__builtin_amdgcn_tensor_load_to_lds)
  if ((threadIdx.x >> 5) == 0) {
    tdm_load_16x1024(sA, A + (size_t)m0 * H_);
    __builtin_amdgcn_s_wait_tensorcnt(0);
  }
#else
  coop_stage_16x1024(sA, A + (size_t)m0 * H_);
#endif
  __syncthreads();

  const int wid  = threadIdx.x >> 5;
  const int n0   = (blockIdx.y * 8 + wid) << 5;   // 2 tiles: n0, n0+16
  const int lane = threadIdx.x & 31;

  v8f acc0 = {}, acc1 = {};

  v16bf b0 = load_b32x16(Wf, n0,      H_, 0);
  v16bf b1 = load_b32x16(Wf, n0 + 16, H_, 0);
  for (int k0 = 0; k0 < H_; k0 += 32) {
    v16bf ax = lds_a16x32(sA, k0);
    const int kn = (k0 + 32) & (H_ - 1);
    v16bf nb0 = load_b32x16(Wf, n0,      H_, kn);
    v16bf nb1 = load_b32x16(Wf, n0 + 16, H_, kn);
    acc0 = wmma_bf16(ax, b0, acc0);
    acc1 = wmma_bf16(ax, b1, acc1);
    b0 = nb0;
    b1 = nb1;
  }

  const int   n   = lane & 15;
  const int   mb  = (lane & 16) ? 8 : 0;
  const float bv0 = fb[n0 + n];
  const float bv1 = fb[n0 + 16 + n];
#pragma unroll
  for (int r = 0; r < 8; ++r) {
    size_t row = (size_t)(m0 + mb + r) * V_;
    // 524 MB write stream: non-temporal, keep L2 for fc_W
    __builtin_nontemporal_store(acc0[r] + bv0, &out[row + n0 + n]);
    __builtin_nontemporal_store(acc1[r] + bv1, &out[row + n0 + 16 + n]);
  }
}

// ---------- elementwise utility kernels ----------

__global__ void f32_to_bf16_kernel(const float* __restrict__ in,
                                   bf16_t* __restrict__ out, size_t n) {
  size_t i = (size_t)blockIdx.x * blockDim.x + threadIdx.x;
  if (i < n) out[i] = f2bf(in[i]);
}

__global__ void add_bias_kernel(const float* __restrict__ a,
                                const float* __restrict__ b,
                                float* __restrict__ out, int n) {
  int i = blockIdx.x * blockDim.x + threadIdx.x;
  if (i < n) out[i] = a[i] + b[i];
}

__global__ void copy_f32_kernel(const float* __restrict__ in,
                                float* __restrict__ out, size_t n) {
  size_t i = (size_t)blockIdx.x * blockDim.x + threadIdx.x;
  if (i < n) out[i] = in[i];
}

__global__ void embed_kernel(const int* __restrict__ x,
                             const float* __restrict__ emb,
                             bf16_t* __restrict__ y, size_t n) {
  size_t i = (size_t)blockIdx.x * blockDim.x + threadIdx.x;
  if (i < n) {
    size_t sb = i / H_;
    int    k  = (int)(i % H_);
    int    tok = x[sb];
    y[i] = f2bf(emb[(size_t)tok * H_ + k]);
  }
}

// ---------- host launcher ----------

extern "C" void kernel_launch(void* const* d_in, const int* in_sizes, int n_in,
                              void* d_out, int out_size, void* d_ws, size_t ws_size,
                              hipStream_t stream) {
  (void)in_sizes; (void)n_in; (void)out_size; (void)ws_size;

  const int*   x    = (const int*)  d_in[0];
  const float* h0   = (const float*)d_in[1];
  const float* c0   = (const float*)d_in[2];
  const float* emb  = (const float*)d_in[3];
  const float* Wih  = (const float*)d_in[4];   // [L,4H,H]
  const float* Whh  = (const float*)d_in[5];   // [L,4H,H]
  const float* bih  = (const float*)d_in[6];   // [L,4H]
  const float* bhh  = (const float*)d_in[7];   // [L,4H]
  const float* fcW  = (const float*)d_in[8];   // [V,H]
  const float* fcb  = (const float*)d_in[9];   // [V]
  float* out = (float*)d_out;

  // -------- carve workspace --------
  char* ws = (char*)d_ws;
  auto carve = [&](size_t bytes) -> char* {
    char* p = ws;
    ws += (bytes + 255) & ~(size_t)255;
    return p;
  };
  const size_t WSZ = (size_t)L_ * 4 * H_ * H_;
  const size_t YSZ = (size_t)S_ * B_ * H_;
  const size_t HSZ = (size_t)L_ * B_ * H_;

  bf16_t* WihB  = (bf16_t*)carve(WSZ * sizeof(bf16_t));
  bf16_t* WhhB  = (bf16_t*)carve(WSZ * sizeof(bf16_t));
  bf16_t* fcWB  = (bf16_t*)carve((size_t)V_ * H_ * sizeof(bf16_t));
  float*  biasW = (float*) carve((size_t)L_ * 4 * H_ * sizeof(float));
  bf16_t* ybufA = (bf16_t*)carve(YSZ * sizeof(bf16_t));
  bf16_t* ybufB = (bf16_t*)carve(YSZ * sizeof(bf16_t));
  bf16_t* h0B   = (bf16_t*)carve(HSZ * sizeof(bf16_t));
  float*  cst   = (float*) carve(HSZ * sizeof(float));
  float*  hst   = (float*) carve(HSZ * sizeof(float));

  auto nblk = [](size_t n) { return (unsigned)((n + 255) / 256); };

  // -------- one-time conversions / state init --------
  f32_to_bf16_kernel<<<nblk(WSZ), 256, 0, stream>>>(Wih, WihB, WSZ);
  f32_to_bf16_kernel<<<nblk(WSZ), 256, 0, stream>>>(Whh, WhhB, WSZ);
  f32_to_bf16_kernel<<<nblk((size_t)V_ * H_), 256, 0, stream>>>(fcW, fcWB, (size_t)V_ * H_);
  f32_to_bf16_kernel<<<nblk(HSZ), 256, 0, stream>>>(h0, h0B, HSZ);
  add_bias_kernel<<<nblk((size_t)L_ * 4 * H_), 256, 0, stream>>>(bih, bhh, biasW, L_ * 4 * H_);
  copy_f32_kernel<<<nblk(HSZ), 256, 0, stream>>>(c0, cst, HSZ);
  copy_f32_kernel<<<nblk(HSZ), 256, 0, stream>>>(h0, hst, HSZ);

  // -------- embedding gather -> bf16 sequence buffer --------
  embed_kernel<<<nblk(YSZ), 256, 0, stream>>>(x, emb, ybufA, YSZ);

  // -------- LSTM layers (time recurrence = sequential launches) --------
  bf16_t* ybufs[2] = {ybufA, ybufB};
  for (int l = 0; l < L_; ++l) {
    const bf16_t* yin  = ybufs[l & 1];
    bf16_t*       yout = ybufs[1 - (l & 1)];
    const bf16_t* WihL = WihB + (size_t)l * 4 * H_ * H_;
    const bf16_t* WhhL = WhhB + (size_t)l * 4 * H_ * H_;
    const float*  bL   = biasW + (size_t)l * 4 * H_;
    float*        cL   = cst + (size_t)l * B_ * H_;
    float*        hL   = hst + (size_t)l * B_ * H_;
    for (int t = 0; t < S_; ++t) {
      const bf16_t* xin = yin + (size_t)t * B_ * H_;
      const bf16_t* hp  = (t == 0) ? (h0B + (size_t)l * B_ * H_)
                                   : (yout + (size_t)(t - 1) * B_ * H_);
      lstm_cell_kernel<<<dim3(8), dim3(256), 0, stream>>>(
          xin, hp, WihL, WhhL, bL, cL, hL, yout + (size_t)t * B_ * H_);
    }
  }
  const bf16_t* yfin = ybufs[0];   // l=3 wrote into ybufA

  // -------- FC GEMM: [4096,1024] x [1024,32000] --------
  fc_gemm_kernel<<<dim3((S_ * B_) / 16, (V_ / 16) / 2 / 8), dim3(256), 0, stream>>>(
      yfin, fcWB, fcb, out);

  // -------- pack h_fin / c_fin after logits --------
  float* out_h = out + (size_t)S_ * B_ * V_;
  float* out_c = out_h + HSZ;
  copy_f32_kernel<<<nblk(HSZ), 256, 0, stream>>>(hst, out_h, HSZ);
  copy_f32_kernel<<<nblk(HSZ), 256, 0, stream>>>(cst, out_c, HSZ);
}
